// graphcl_82145544503554
// MI455X (gfx1250) — matrix operand order repaired and checked by reading, not compiled, not run on hardware
//
#include <hip/hip_runtime.h>

// RReLU eval-mode slope: (1/8 + 1/3)/2
#define SLOPE_RRELU 0.2291666666666667f

typedef _Float16 f16;
typedef _Float16 v8h  __attribute__((ext_vector_type(8)));
typedef _Float16 v16h __attribute__((ext_vector_type(16)));
typedef float    v8f  __attribute__((ext_vector_type(8)));
typedef int      v4i  __attribute__((vector_size(16)));   // gcc-style, matches builtin proto

#if defined(__HIP_DEVICE_COMPILE__) && __has_builtin(__builtin_amdgcn_global_load_async_to_lds_b128)
#define ASYNC_LDS 1
typedef __attribute__((address_space(1))) v4i* as1_v4i_ptr;
typedef __attribute__((address_space(3))) v4i* as3_v4i_ptr;
#else
#define ASYNC_LDS 0
#endif

// ---------------------------------------------------------------------------
// Edge scatter-add: agg[dst] += h[src], float4 vectorized, C divisible by 4.
// ---------------------------------------------------------------------------
__global__ void scatter_add_kernel(const float* __restrict__ h,
                                   const int* __restrict__ src,
                                   const int* __restrict__ dst,
                                   float* __restrict__ agg,
                                   int E, int C) {
    long long t = (long long)blockIdx.x * blockDim.x + threadIdx.x;
    int chunks = C >> 2;
    long long total = (long long)E * chunks;
    if (t >= total) return;
    int e  = (int)(t / chunks);
    int c4 = ((int)(t % chunks)) << 2;
    int s = src[e], d = dst[e];
    const float4 v = *reinterpret_cast<const float4*>(h + (size_t)s * C + c4);
    float* o = agg + (size_t)d * C + c4;
    atomicAdd(o + 0, v.x);
    atomicAdd(o + 1, v.y);
    atomicAdd(o + 2, v.z);
    atomicAdd(o + 3, v.w);
}

// ---------------------------------------------------------------------------
// out16[N x Cp] = f16(h + agg) with zero-padding for col >= C (agg may be null)
// ---------------------------------------------------------------------------
__global__ void add_cvt_f16_kernel(const float* __restrict__ h,
                                   const float* __restrict__ agg,
                                   f16* __restrict__ out,
                                   int N, int C, int Cp) {
    long long t = (long long)blockIdx.x * blockDim.x + threadIdx.x;
    if (t >= (long long)N * Cp) return;
    int row = (int)(t / Cp);
    int col = (int)(t % Cp);
    float v = 0.0f;
    if (col < C) {
        size_t idx = (size_t)row * C + col;
        v = h[idx];
        if (agg) v += agg[idx];
    }
    out[t] = (f16)v;
}

// ---------------------------------------------------------------------------
// Weight convert + pad: out[Rp x Cp] = f16(w[R x C]), zeros outside.
// ---------------------------------------------------------------------------
__global__ void cvt_w_f16_kernel(const float* __restrict__ w, f16* __restrict__ out,
                                 int R, int C, int Rp, int Cp) {
    int t = blockIdx.x * blockDim.x + threadIdx.x;
    if (t >= Rp * Cp) return;
    int r = t / Cp, c = t % Cp;
    out[t] = (r < R && c < C) ? (f16)w[r * C + c] : (f16)0.0f;
}

// ---------------------------------------------------------------------------
// global_add_pool: g[batch[n]][c] += hlin[n][c]
// ---------------------------------------------------------------------------
__global__ void pool_kernel(const float* __restrict__ hlin,
                            const int* __restrict__ batch,
                            float* __restrict__ g, int N, int C) {
    long long t = (long long)blockIdx.x * blockDim.x + threadIdx.x;
    if (t >= (long long)N * C) return;
    int n = (int)(t / C);
    int c = (int)(t % C);
    atomicAdd(&g[(size_t)batch[n] * C + c], hlin[t]);
}

// ---------------------------------------------------------------------------
// LayerNorm over C=120, emit f16 padded to Cp=128 (blockDim == Cp == 128)
// ---------------------------------------------------------------------------
__global__ void layernorm_f16_kernel(const float* __restrict__ g,
                                     const float* __restrict__ gamma,
                                     const float* __restrict__ beta,
                                     f16* __restrict__ out, int C, int Cp) {
    __shared__ float red[128];
    int b = blockIdx.x, t = threadIdx.x;
    float v = (t < C) ? g[(size_t)b * C + t] : 0.0f;
    red[t] = v; __syncthreads();
    for (int s = 64; s > 0; s >>= 1) { if (t < s) red[t] += red[t + s]; __syncthreads(); }
    float mu = red[0] / C;
    __syncthreads();
    float d = (t < C) ? (v - mu) : 0.0f;
    red[t] = d * d; __syncthreads();
    for (int s = 64; s > 0; s >>= 1) { if (t < s) red[t] += red[t + s]; __syncthreads(); }
    float inv = rsqrtf(red[0] / C + 1e-5f);
    float r = (t < C) ? ((v - mu) * inv * gamma[t] + beta[t]) : 0.0f;
    out[(size_t)b * Cp + t] = (f16)r;
}

// ---------------------------------------------------------------------------
// WMMA GEMM with LDS-staged, double-buffered B panel.
//   Out[M x Nreal] = act(A[M x K] * B[K x Npad] + bias)
//   Block: 256 threads = 8 waves; block tile 128(M) x 128(N);
//   wave tile 16 x 128 (8 accumulator tiles). K staged in BK=64 chunks.
//   B fragment for tile nt is read from LDS into one v16h right before WMMA nt.
//   ACT: 0 = none, 1 = ReLU, 2 = RReLU.  OUT16: write f16, else f32.
// ---------------------------------------------------------------------------
#define GEMM_BK   64
#define GEMM_BSTR 136   // padded LDS row stride in f16 (272B: 16B aligned, low conflict)

template <int ACT, bool OUT16>
__global__ void __launch_bounds__(256)
gemm_wmma_kernel(const f16* __restrict__ A,
                 const f16* __restrict__ B,
                 const float* __restrict__ bias,
                 void* __restrict__ Out,
                 int M, int K, int Npad, int Nreal) {
    __shared__ f16 bs[2][GEMM_BK * GEMM_BSTR];

    const int tid  = threadIdx.x;
    const int lane = tid & 31;
    const int wave = tid >> 5;
    const int mrow0 = blockIdx.x * 128 + wave * 16;
    const int ncol0 = blockIdx.y * 128;

    // A fragment addressing (16-bit A 16x32 layout):
    //   lanes 0-15  : row M = lane,    K = k0+0..7  and k0+16..23
    //   lanes 16-31 : row M = lane-16, K = k0+8..15 and k0+24..31
    // Clamp row for loads (uniform control flow); stores are guarded.
    int arow = mrow0 + (lane & 15);
    if (arow >= M) arow = M - 1;
    const int kA = (lane >> 4) << 3;
    const f16* Abase = A + (size_t)arow * K + kA;

    // ---- stage BK rows of the B panel into LDS buffer `buf` ----
    auto stage = [&](int kb, int buf) {
        // BK rows x 128 cols, copied in 16B units: BK*16 = 1024 units / 256 thr
#pragma unroll
        for (int u = 0; u < 4; ++u) {
            int unit = tid + u * 256;
            int row  = unit >> 4;
            int seg  = unit & 15;
            const f16* gsrc = B + (size_t)(kb + row) * Npad + ncol0 + seg * 8;
            f16* ldst = &bs[buf][row * GEMM_BSTR + seg * 8];
#if ASYNC_LDS
            __builtin_amdgcn_global_load_async_to_lds_b128(
                (as1_v4i_ptr)(size_t)gsrc,
                (as3_v4i_ptr)(size_t)(unsigned)(size_t)ldst,
                0, 0);
#else
            *reinterpret_cast<v8h*>(ldst) = *reinterpret_cast<const v8h*>(gsrc);
#endif
        }
    };

    v8f acc[8] = {};

    stage(0, 0);
#if ASYNC_LDS
    asm volatile("s_wait_asynccnt 0" ::: "memory");
#endif
    __syncthreads();

    for (int kb = 0; kb < K; kb += GEMM_BK) {
        const int cur = (kb >> 6) & 1;
        const bool more = (kb + GEMM_BK) < K;
        if (more) stage(kb + GEMM_BK, cur ^ 1);

#pragma unroll
        for (int half = 0; half < 2; ++half) {
            const int kc = half * 32;
            v8h alo = *reinterpret_cast<const v8h*>(Abase + kb + kc);
            v8h ahi = *reinterpret_cast<const v8h*>(Abase + kb + kc + 16);
            v16h a = __builtin_shufflevector(alo, ahi, 0, 1, 2, 3, 4, 5, 6, 7,
                                                       8, 9, 10, 11, 12, 13, 14, 15);
            // B fragment: lane L holds N col (L & 15); K = kc+kA+0..7 and kc+16+kA+0..7 (same as A).
            const f16* bcol = &bs[cur][(kc + kA) * GEMM_BSTR + (lane & 15)];
#pragma unroll
            for (int nt = 0; nt < 8; ++nt) {
                v16h b;
#pragma unroll
                for (int s = 0; s < 8; ++s) {
                    b[s]     = bcol[s * GEMM_BSTR + nt * 16];
                    b[s + 8] = bcol[(16 + s) * GEMM_BSTR + nt * 16];
                }
                acc[nt] = __builtin_amdgcn_wmma_f32_16x16x32_f16(
                    false, a, false, b, (short)0, acc[nt], false, false);
            }
        }

#if ASYNC_LDS
        if (more) asm volatile("s_wait_asynccnt 0" ::: "memory");
#endif
        __syncthreads();
    }

    // C/D layout: VGPR r -> lanes 0-15: M=r, N=lane; lanes 16-31: M=r+8, N=lane-16
    const int r0   = (lane >> 4) << 3;
    const int coll = lane & 15;
#pragma unroll
    for (int nt = 0; nt < 8; ++nt) {
        int col = ncol0 + nt * 16 + coll;
        if (col >= Nreal) continue;
        float bv = bias ? bias[col] : 0.0f;
#pragma unroll
        for (int r = 0; r < 8; ++r) {
            int row = mrow0 + r0 + r;
            if (row >= M) continue;
            float v = acc[nt][r] + bv;
            if (ACT == 1) v = fmaxf(v, 0.0f);
            if (ACT == 2) v = (v >= 0.0f) ? v : SLOPE_RRELU * v;
            if (OUT16)
                reinterpret_cast<f16*>(Out)[(size_t)row * Nreal + col] = (f16)v;
            else
                reinterpret_cast<float*>(Out)[(size_t)row * Nreal + col] = v;
        }
    }
}

// ---------------------------------------------------------------------------
extern "C" void kernel_launch(void* const* d_in, const int* in_sizes, int n_in,
                              void* d_out, int out_size, void* d_ws, size_t ws_size,
                              hipStream_t stream) {
    const float* x     = (const float*)d_in[0];
    const int*   ei    = (const int*)d_in[1];
    const int*   batch = (const int*)d_in[2];

    const int N   = in_sizes[2];
    const int E   = in_sizes[1] / 2;
    const int Fin = in_sizes[0] / N;   // 108
    const int H = 256, O = 120, G = 256;
    const int FinP = 128, OP = 128;    // K / N padding to WMMA-friendly sizes

    const int* src = ei;
    const int* dst = ei + E;

    // ---- workspace carve ----
    char* ws = (char*)d_ws;
    size_t off = 0;
    auto carve = [&](size_t bytes) -> void* {
        void* p = ws + off;
        off += (bytes + 255) & ~(size_t)255;
        return p;
    };
    float* hcur = (float*)carve((size_t)N * H * 4);
    float* agg  = (float*)carve((size_t)N * H * 4);
    f16*   a16  = (f16*)carve((size_t)N * H * 2);
    f16*   t16  = (f16*)carve((size_t)N * H * 2);
    float* hlin = agg;  // agg is dead by the time hlin is produced
    float* gbuf = (float*)carve((size_t)G * O * 4);
    f16* hn16   = (f16*)carve((size_t)G * OP * 2);
    f16* z16    = (f16*)carve((size_t)G * H * 2);
    f16* w16[10];
    w16[0] = (f16*)carve((size_t)FinP * H * 2);               // w1a padded 128x256
    for (int i = 1; i < 10; ++i) w16[i] = (f16*)carve((size_t)H * H * 2);
    f16* linw16 = (f16*)carve((size_t)H * OP * 2);            // 256x128 (cols padded)
    f16* pw1_16 = (f16*)carve((size_t)OP * H * 2);            // 128x256 (rows padded)
    f16* pw2_16 = (f16*)carve((size_t)H * OP * 2);            // 256x128 (cols padded)

    // ---- weight conversion to f16 (cheap; deterministic per launch) ----
    auto cvtW = [&](const float* s, f16* d, int R, int C, int Rp, int Cp) {
        int tot = Rp * Cp;
        cvt_w_f16_kernel<<<(tot + 255) / 256, 256, 0, stream>>>(s, d, R, C, Rp, Cp);
    };
    cvtW((const float*)d_in[3], w16[0], Fin, H, FinP, H);     // w1a
    cvtW((const float*)d_in[5], w16[1], H, H, H, H);          // w1b
    for (int l = 2; l <= 5; ++l) {
        cvtW((const float*)d_in[3 + 4 * (l - 1)], w16[2 * (l - 1)],     H, H, H, H);
        cvtW((const float*)d_in[5 + 4 * (l - 1)], w16[2 * (l - 1) + 1], H, H, H, H);
    }
    cvtW((const float*)d_in[23], linw16, H, O, H, OP);
    cvtW((const float*)d_in[27], pw1_16, O, H, OP, H);
    cvtW((const float*)d_in[29], pw2_16, H, O, H, OP);

    // ---- GEMM launcher ----
    auto gemm = [&](const f16* A, const f16* B, const float* bias, void* Out,
                    int M, int K, int Npad, int Nreal, int act, bool out16) {
        dim3 grid((M + 127) / 128, Npad / 128);
        if (act == 1 && out16)
            gemm_wmma_kernel<1, true ><<<grid, 256, 0, stream>>>(A, B, bias, Out, M, K, Npad, Nreal);
        else if (act == 2 && out16)
            gemm_wmma_kernel<2, true ><<<grid, 256, 0, stream>>>(A, B, bias, Out, M, K, Npad, Nreal);
        else if (act == 2)
            gemm_wmma_kernel<2, false><<<grid, 256, 0, stream>>>(A, B, bias, Out, M, K, Npad, Nreal);
        else
            gemm_wmma_kernel<0, false><<<grid, 256, 0, stream>>>(A, B, bias, Out, M, K, Npad, Nreal);
    };

    // ================= Layer 1 (K = 108 -> padded 128) =================
    (void)hipMemsetAsync(agg, 0, (size_t)N * Fin * 4, stream);
    {
        long long tot = (long long)E * (Fin >> 2);
        scatter_add_kernel<<<(unsigned)((tot + 255) / 256), 256, 0, stream>>>(x, src, dst, agg, E, Fin);
    }
    {
        long long tot = (long long)N * FinP;
        add_cvt_f16_kernel<<<(unsigned)((tot + 255) / 256), 256, 0, stream>>>(x, agg, a16, N, Fin, FinP);
    }
    gemm(a16, w16[0], (const float*)d_in[4], t16,  N, FinP, H, H, /*ReLU */1, true);
    gemm(t16, w16[1], (const float*)d_in[6], hcur, N, H,    H, H, /*RReLU*/2, false);

    // ================= Layers 2..5 =================
    for (int l = 2; l <= 5; ++l) {
        const float* ba = (const float*)d_in[4 + 4 * (l - 1)];
        const float* bb = (const float*)d_in[6 + 4 * (l - 1)];
        (void)hipMemsetAsync(agg, 0, (size_t)N * H * 4, stream);
        {
            long long tot = (long long)E * (H >> 2);
            scatter_add_kernel<<<(unsigned)((tot + 255) / 256), 256, 0, stream>>>(hcur, src, dst, agg, E, H);
        }
        {
            long long tot = (long long)N * H;
            add_cvt_f16_kernel<<<(unsigned)((tot + 255) / 256), 256, 0, stream>>>(hcur, agg, a16, N, H, H);
        }
        gemm(a16, w16[2 * (l - 1)],     ba, t16,  N, H, H, H, 1, true);
        gemm(t16, w16[2 * (l - 1) + 1], bb, hcur, N, H, H, H, 2, false);
    }

    // ================= Linear head: hcur @ lin_w + lin_b -> hlin [N x 120] ===
    {
        long long tot = (long long)N * H;
        add_cvt_f16_kernel<<<(unsigned)((tot + 255) / 256), 256, 0, stream>>>(hcur, nullptr, a16, N, H, H);
    }
    gemm(a16, linw16, (const float*)d_in[24], hlin, N, H, OP, O, 0, false);

    // ================= global_add_pool -> gbuf [G x 120] =================
    (void)hipMemsetAsync(gbuf, 0, (size_t)G * O * 4, stream);
    {
        long long tot = (long long)N * O;
        pool_kernel<<<(unsigned)((tot + 255) / 256), 256, 0, stream>>>(hlin, batch, gbuf, N, O);
    }

    // ================= LayerNorm -> hn16 [G x 128 f16] =================
    layernorm_f16_kernel<<<G, 128, 0, stream>>>(
        gbuf, (const float*)d_in[25], (const float*)d_in[26], hn16, O, OP);

    // ================= Projection head =================
    gemm(hn16, pw1_16, (const float*)d_in[28], z16,   G, OP, H,  H, 2, true);
    gemm(z16,  pw2_16, (const float*)d_in[30], d_out, G, H,  OP, O, 0, false);
}
